// GraphNCF_A_13426067768108
// MI455X (gfx1250) — matrix operand order repaired and checked by reading, not compiled, not run on hardware
//
#include <hip/hip_runtime.h>
#include <hip/hip_bf16.h>

#define N_USERS   50000
#define N_ITEMS   50000
#define N_NODES   (N_USERS + N_ITEMS)
#define EMBED     64
#define HEADS     4
#define HEAD_DIM  16
#define LAYERS    3
#define N_EDGES   1600000
#define BATCH     4096
#define RED_BLOCKS 1024
#define PACKW_ELEMS 4096          // 8 fragments * 32 lanes * 16 halfs

typedef __attribute__((ext_vector_type(16))) _Float16 v16h;
typedef __attribute__((ext_vector_type(8)))  float    v8f;

// ---------------------------------------------------------------------------
// Layer-0 init: prev = concat(user_emb, item_emb); also all_emb[:, 0:64]
// ---------------------------------------------------------------------------
__global__ __launch_bounds__(256) void init_emb_kernel(
    const float* __restrict__ ue, const float* __restrict__ ie,
    float* __restrict__ prev, float* __restrict__ all_emb)
{
    int idx = blockIdx.x * 256 + threadIdx.x;        // N_NODES*64 threads
    if (idx >= N_NODES * EMBED) return;
    int row = idx >> 6, c = idx & 63;
    float v = (row < N_USERS) ? ue[row * EMBED + c]
                              : ie[(row - N_USERS) * EMBED + c];
    prev[idx] = v;
    all_emb[(size_t)row * (EMBED * (LAYERS + 1)) + c] = v;
}

// ---------------------------------------------------------------------------
// Elementwise helpers
// ---------------------------------------------------------------------------
__global__ __launch_bounds__(256) void square_kernel(
    const float* __restrict__ x, float* __restrict__ y, int n)
{
    int i = blockIdx.x * 256 + threadIdx.x;
    if (i < n) { float v = x[i]; y[i] = v * v; }
}

__global__ __launch_bounds__(256) void copy_kernel(
    const float* __restrict__ x, float* __restrict__ y, int n)
{
    int i = blockIdx.x * 256 + threadIdx.x;
    if (i < n) y[i] = x[i];
}

__global__ __launch_bounds__(256) void zero_kernel(float* __restrict__ y, int n)
{
    int i = blockIdx.x * 256 + threadIdx.x;
    if (i < n) y[i] = 0.0f;
}

// ---------------------------------------------------------------------------
// Pre-pack a 64x64 weight into WMMA B-fragment f16 layout.
//   fragment tile = s*4 + t (s = K-step 0..1, t = col-tile 0..3)
//   per lane (hi = lane>>4, lrow = lane&15): col j = 16t+lrow,
//   K = 32s + 16*hi + kk, kk = 0..15 contiguous halfs.
//   transB=1 : Beff[k,j] = W[j,k];  transB=0 : Beff[k,j] = W[k,j]
// ---------------------------------------------------------------------------
__global__ __launch_bounds__(256) void pack_weight_kernel(
    const float* __restrict__ W, _Float16* __restrict__ out, int transB)
{
    int idx = blockIdx.x * 256 + threadIdx.x;        // PACKW_ELEMS threads
    if (idx >= PACKW_ELEMS) return;
    int kk   = idx & 15;
    int lane = (idx >> 4) & 31;
    int tile = idx >> 9;                             // 0..7
    int s = tile >> 2, t = tile & 3;
    int hi = lane >> 4, lrow = lane & 15;
    int j = 16 * t + lrow;
    int k = 32 * s + 16 * hi + kk;
    float w = transB ? W[j * EMBED + k] : W[k * EMBED + j];
    out[idx] = (_Float16)w;
}

// ---------------------------------------------------------------------------
// sp_mm scatter: out[src] += w[e] * xin[dst]   (out pre-initialized to xin)
// 16 lanes per edge, float4 per lane.
// ---------------------------------------------------------------------------
__global__ __launch_bounds__(256) void spmm_scatter_kernel(
    const int* __restrict__ esrc, const int* __restrict__ edst,
    const float* __restrict__ w, const float* __restrict__ xin,
    float* __restrict__ out)
{
    int tid = blockIdx.x * 256 + threadIdx.x;        // N_EDGES*16 threads
    int e = tid >> 4;
    if (e >= N_EDGES) return;
    int sub = tid & 15;
    int src = esrc[e], dst = edst[e];
    float ww = w[e];
    float4 m = *(const float4*)(xin + (size_t)dst * EMBED + sub * 4);
    float* o = out + (size_t)src * EMBED + sub * 4;
    atomicAdd(o + 0, ww * m.x);
    atomicAdd(o + 1, ww * m.y);
    atomicAdd(o + 2, ww * m.z);
    atomicAdd(o + 3, ww * m.w);
}

// ---------------------------------------------------------------------------
// Dense GEMM via WMMA:  C[n,j] = sum_k A[n,k]*Beff[k,j] (+ bias[j])
// B is pre-packed into fragment layout (PW). 128 threads = 4 waves; each
// wave: 16 rows x 64 cols via 4 col-tiles x 2 K-steps of
// v_wmma_f32_16x16x32_f16 (f32 accumulate). Stores take a wave-uniform
// fast path (unguarded, lane-coalesced) except in the single ragged block.
// ---------------------------------------------------------------------------
__global__ __launch_bounds__(128) void gemm64_wmma_kernel(
    const float* __restrict__ A, const _Float16* __restrict__ PW,
    const float* __restrict__ bias, float* __restrict__ C, int N)
{
    const int wave = threadIdx.x >> 5;
    const int lane = threadIdx.x & 31;
    const int lrow = lane & 15;
    const int hi   = lane >> 4;
    const int rowBase = (blockIdx.x * 4 + wave) * 16;

    int row  = rowBase + lrow;
    int rowc = row < N ? row : N - 1;                 // clamp, keep EXEC full
    const float* arow = A + (size_t)rowc * EMBED;

    // A fragments for K-tiles s=0,1 (contiguous k = 32s + 16*hi + e, same order as packed B)
    v16h afrag[2];
    #pragma unroll
    for (int s = 0; s < 2; ++s) {
        #pragma unroll
        for (int p = 0; p < 8; ++p) {
            int e = 2 * p;
            int k = e + (hi << 4) + 32 * s;
            float2 f = *(const float2*)(arow + k);
            afrag[s][e]     = (_Float16)f.x;
            afrag[s][e + 1] = (_Float16)f.y;
        }
    }

    const bool fullTile = (rowBase + 16) <= N;        // wave-uniform

    #pragma unroll
    for (int t = 0; t < 4; ++t) {                     // 4 output col-tiles
        int j = 16 * t + lrow;
        v8f acc = {0.f, 0.f, 0.f, 0.f, 0.f, 0.f, 0.f, 0.f};
        #pragma unroll
        for (int s = 0; s < 2; ++s) {                 // 2 K-steps of 32
            v16h b = *(const v16h*)(PW + (((s << 2) | t) * 32 + lane) * 16);
            acc = __builtin_amdgcn_wmma_f32_16x16x32_f16(
                false, afrag[s], false, b, (short)0, acc, false, false);
        }
        float bv = bias ? bias[j] : 0.0f;
        if (fullTile) {                               // common case: no guards
            float* cb = C + (size_t)(rowBase + 8 * hi) * EMBED + j;
            #pragma unroll
            for (int r = 0; r < 8; ++r)               // C/D: row = r + 8*hi
                cb[(size_t)r * EMBED] = acc[r] + bv;
        } else {                                      // ragged tail block
            #pragma unroll
            for (int r = 0; r < 8; ++r) {
                int orow = rowBase + r + 8 * hi;
                if (orow < N) C[(size_t)orow * EMBED + j] = acc[r] + bv;
            }
        }
    }
}

// ---------------------------------------------------------------------------
// Per-edge attention scores: scores[e,h] = dot16(q[src,h], k[dst,h]) / 4
// ---------------------------------------------------------------------------
__global__ __launch_bounds__(256) void edge_scores_kernel(
    const int* __restrict__ esrc, const int* __restrict__ edst,
    const float* __restrict__ q, const float* __restrict__ k,
    float* __restrict__ scores)
{
    int tid = blockIdx.x * 256 + threadIdx.x;         // N_EDGES*HEADS
    int e = tid >> 2;
    if (e >= N_EDGES) return;
    int h = tid & 3;
    const float4* qq = (const float4*)(q + (size_t)esrc[e] * EMBED + h * HEAD_DIM);
    const float4* kk = (const float4*)(k + (size_t)edst[e] * EMBED + h * HEAD_DIM);
    float s = 0.0f;
    #pragma unroll
    for (int i = 0; i < 4; ++i) {
        float4 a = qq[i], b = kk[i];
        s += a.x * b.x + a.y * b.y + a.z * b.z + a.w * b.w;
    }
    scores[(size_t)e * HEADS + h] = s * 0.25f;        // 1/sqrt(16)
}

// ---------------------------------------------------------------------------
// Global (per-head) softmax reductions: hierarchical max, then exp-sum
// ---------------------------------------------------------------------------
__global__ __launch_bounds__(256) void head_max_partial_kernel(
    const float* __restrict__ scores, float* __restrict__ partial)
{
    int h = blockIdx.y;
    float m = -3.0e38f;
    for (int e = blockIdx.x * 256 + threadIdx.x; e < N_EDGES; e += gridDim.x * 256)
        m = fmaxf(m, scores[(size_t)e * HEADS + h]);
    __shared__ float sm[256];
    sm[threadIdx.x] = m; __syncthreads();
    for (int s = 128; s > 0; s >>= 1) {
        if (threadIdx.x < s) sm[threadIdx.x] = fmaxf(sm[threadIdx.x], sm[threadIdx.x + s]);
        __syncthreads();
    }
    if (threadIdx.x == 0) partial[h * gridDim.x + blockIdx.x] = sm[0];
}

__global__ __launch_bounds__(256) void head_sum_partial_kernel(
    const float* __restrict__ scores, const float* __restrict__ headmax,
    float* __restrict__ partial)
{
    int h = blockIdx.y;
    float mx = headmax[h];
    float acc = 0.0f;
    for (int e = blockIdx.x * 256 + threadIdx.x; e < N_EDGES; e += gridDim.x * 256)
        acc += __expf(scores[(size_t)e * HEADS + h] - mx);
    __shared__ float sm[256];
    sm[threadIdx.x] = acc; __syncthreads();
    for (int s = 128; s > 0; s >>= 1) {
        if (threadIdx.x < s) sm[threadIdx.x] += sm[threadIdx.x + s];
        __syncthreads();
    }
    if (threadIdx.x == 0) partial[h * gridDim.x + blockIdx.x] = sm[0];
}

__global__ __launch_bounds__(256) void final_reduce_kernel(
    const float* __restrict__ partial, float* __restrict__ out, int n, int isMax)
{
    int h = blockIdx.x;
    float acc = isMax ? -3.0e38f : 0.0f;
    for (int i = threadIdx.x; i < n; i += 256) {
        float v = partial[h * n + i];
        acc = isMax ? fmaxf(acc, v) : acc + v;
    }
    __shared__ float sm[256];
    sm[threadIdx.x] = acc; __syncthreads();
    for (int s = 128; s > 0; s >>= 1) {
        if (threadIdx.x < s)
            sm[threadIdx.x] = isMax ? fmaxf(sm[threadIdx.x], sm[threadIdx.x + s])
                                    : sm[threadIdx.x] + sm[threadIdx.x + s];
        __syncthreads();
    }
    if (threadIdx.x == 0) out[h] = sm[0];
}

// ---------------------------------------------------------------------------
// Attention weighted scatter: out[src,h,:] += attn(e,h) * v[dst,h,:]
// ---------------------------------------------------------------------------
__global__ __launch_bounds__(256) void attn_scatter_kernel(
    const int* __restrict__ esrc, const int* __restrict__ edst,
    const float* __restrict__ scores, const float* __restrict__ headmax,
    const float* __restrict__ headsum, const float* __restrict__ v,
    float* __restrict__ out)
{
    int tid = blockIdx.x * 256 + threadIdx.x;         // N_EDGES*16
    int e = tid >> 4;
    if (e >= N_EDGES) return;
    int sub = tid & 15;
    int h = sub >> 2;
    int d4 = (sub & 3) * 4;
    float a = __expf(scores[(size_t)e * HEADS + h] - headmax[h]) / headsum[h];
    float4 vv = *(const float4*)(v + (size_t)edst[e] * EMBED + h * HEAD_DIM + d4);
    float* o = out + (size_t)esrc[e] * EMBED + h * HEAD_DIM + d4;
    atomicAdd(o + 0, a * vv.x);
    atomicAdd(o + 1, a * vv.y);
    atomicAdd(o + 2, a * vv.z);
    atomicAdd(o + 3, a * vv.w);
}

// ---------------------------------------------------------------------------
// Combine: prev = normalize(leaky_relu(t1+t2, 0.2)); also write all_emb block.
// One wave32 per row; 2 floats/lane; wave reduction for the L2 norm.
// ---------------------------------------------------------------------------
__global__ __launch_bounds__(256) void combine_norm_kernel(
    const float* __restrict__ t1, const float* __restrict__ t2,
    float* __restrict__ prev, float* __restrict__ all_emb, int layerCol)
{
    int wave = threadIdx.x >> 5;
    int lane = threadIdx.x & 31;
    int row = blockIdx.x * 8 + wave;
    if (row >= N_NODES) return;
    size_t off = (size_t)row * EMBED + lane * 2;
    float2 a = *(const float2*)(t1 + off);
    float2 b = *(const float2*)(t2 + off);
    float x0 = a.x + b.x, x1 = a.y + b.y;
    x0 = x0 > 0.0f ? x0 : 0.2f * x0;
    x1 = x1 > 0.0f ? x1 : 0.2f * x1;
    float ss = x0 * x0 + x1 * x1;
    #pragma unroll
    for (int o = 16; o > 0; o >>= 1) ss += __shfl_xor(ss, o, 32);
    float inv = 1.0f / fmaxf(sqrtf(ss), 1e-12f);
    x0 *= inv; x1 *= inv;
    prev[off]     = x0;
    prev[off + 1] = x1;
    size_t ao = (size_t)row * (EMBED * (LAYERS + 1)) + layerCol + lane * 2;
    all_emb[ao]     = x0;
    all_emb[ao + 1] = x1;
}

// ---------------------------------------------------------------------------
// Final gather: (user_final[users], item_final[pos], item_final[neg]) flat.
// ---------------------------------------------------------------------------
__global__ __launch_bounds__(256) void gather_out_kernel(
    const float* __restrict__ all_emb, const int* __restrict__ users,
    const int* __restrict__ pos, const int* __restrict__ neg,
    float* __restrict__ out)
{
    int tid = blockIdx.x * 256 + threadIdx.x;         // 3*BATCH*64 threads
    if (tid >= 3 * BATCH * 64) return;
    int which = tid / (BATCH * 64);
    int r = (tid / 64) % BATCH;
    int sub = tid & 63;
    int node;
    if (which == 0)      node = users[r];
    else if (which == 1) node = N_USERS + pos[r];
    else                 node = N_USERS + neg[r];
    float4 vdat = *(const float4*)(all_emb + (size_t)node * 256 + sub * 4);
    *(float4*)(out + (size_t)which * BATCH * 256 + (size_t)r * 256 + sub * 4) = vdat;
}

// ---------------------------------------------------------------------------
// Host orchestration
// ---------------------------------------------------------------------------
extern "C" void kernel_launch(void* const* d_in, const int* in_sizes, int n_in,
                              void* d_out, int out_size, void* d_ws, size_t ws_size,
                              hipStream_t stream) {
    const float* ue   = (const float*)d_in[0];
    const float* ie   = (const float*)d_in[1];
    const float* Wq   = (const float*)d_in[2];
    const float* Wk   = (const float*)d_in[3];
    const float* Wv   = (const float*)d_in[4];
    const float* Wo   = (const float*)d_in[5];
    const float* bo   = (const float*)d_in[6];
    const float* Wl_a = (const float*)d_in[7];
    const float* bl_a = (const float*)d_in[8];
    const float* ew   = (const float*)d_in[9];
    const int*   eidx = (const int*)d_in[10];
    const int*   users = (const int*)d_in[11];
    const int*   pos   = (const int*)d_in[12];
    const int*   neg   = (const int*)d_in[13];
    const int* esrc = eidx;
    const int* edst = eidx + N_EDGES;
    float* out = (float*)d_out;

    const size_t NE = (size_t)N_NODES * EMBED;        // 6.4M floats
    float* ws = (float*)d_ws;
    float* prev    = ws;                 ws += NE;
    float* xsq     = ws;                 ws += NE;
    float* spout   = ws;                 ws += NE;
    float* q       = ws;                 ws += NE;
    float* k       = ws;                 ws += NE;
    float* v       = ws;                 ws += NE;
    float* attnout = ws;                 ws += NE;
    float* tmp     = ws;                 ws += NE;
    float* t1      = ws;                 ws += NE;
    float* t2      = ws;                 ws += NE;
    float* all_emb = ws;                 ws += (size_t)N_NODES * 256;
    float* scores  = ws;                 ws += (size_t)N_EDGES * HEADS;
    float* partial = ws;                 ws += (size_t)HEADS * RED_BLOCKS;
    float* headmax = ws;                 ws += HEADS;
    float* headsum = ws;                 ws += HEADS;
    ws += 16;                                          // keep 32B alignment pad
    _Float16* pW = (_Float16*)ws;                      // 7 packed weights
    _Float16* pWq = pW + 0 * PACKW_ELEMS;
    _Float16* pWk = pW + 1 * PACKW_ELEMS;
    _Float16* pWv = pW + 2 * PACKW_ELEMS;
    _Float16* pWo = pW + 3 * PACKW_ELEMS;
    _Float16* pWl = pW + 4 * PACKW_ELEMS;              // 3 consecutive layers

    const int eltN    = (int)NE;
    const int eltB    = (eltN + 255) / 256;                 // 25000
    const int edgeB16 = (N_EDGES * 16 + 255) / 256;         // 100000
    const int edgeB4  = (N_EDGES * 4 + 255) / 256;          // 25000
    const int gemmB   = (N_NODES + 63) / 64;                // 1563
    const int combB   = (N_NODES + 7) / 8;                  // 12500
    const int packB   = PACKW_ELEMS / 256;                  // 16

    // One-time (per launch) weight packing into WMMA fragment layout.
    pack_weight_kernel<<<packB, 256, 0, stream>>>(Wq, pWq, 1);
    pack_weight_kernel<<<packB, 256, 0, stream>>>(Wk, pWk, 1);
    pack_weight_kernel<<<packB, 256, 0, stream>>>(Wv, pWv, 1);
    pack_weight_kernel<<<packB, 256, 0, stream>>>(Wo, pWo, 1);
    for (int l = 0; l < LAYERS; ++l)
        pack_weight_kernel<<<packB, 256, 0, stream>>>(
            Wl_a + (size_t)l * EMBED * EMBED, pWl + (size_t)l * PACKW_ELEMS, 0);

    init_emb_kernel<<<eltB, 256, 0, stream>>>(ue, ie, prev, all_emb);

    for (int layer = 0; layer < LAYERS; ++layer) {
        const _Float16* pWlayer = pWl + (size_t)layer * PACKW_ELEMS;
        const float* bl = bl_a + (size_t)layer * EMBED;

        for (int branch = 0; branch < 2; ++branch) {
            const float* xin = prev;
            if (branch == 1) {
                square_kernel<<<eltB, 256, 0, stream>>>(prev, xsq, eltN);
                xin = xsq;
            }
            // sp_mm: out = emb + segment_sum(w * emb[dst])
            copy_kernel<<<eltB, 256, 0, stream>>>(xin, spout, eltN);
            spmm_scatter_kernel<<<edgeB16, 256, 0, stream>>>(esrc, edst, ew, xin, spout);

            // attention
            gemm64_wmma_kernel<<<gemmB, 128, 0, stream>>>(spout, pWq, nullptr, q, N_NODES);
            gemm64_wmma_kernel<<<gemmB, 128, 0, stream>>>(spout, pWk, nullptr, k, N_NODES);
            gemm64_wmma_kernel<<<gemmB, 128, 0, stream>>>(spout, pWv, nullptr, v, N_NODES);
            edge_scores_kernel<<<edgeB4, 256, 0, stream>>>(esrc, edst, q, k, scores);

            head_max_partial_kernel<<<dim3(RED_BLOCKS, HEADS), 256, 0, stream>>>(scores, partial);
            final_reduce_kernel<<<HEADS, 256, 0, stream>>>(partial, headmax, RED_BLOCKS, 1);
            head_sum_partial_kernel<<<dim3(RED_BLOCKS, HEADS), 256, 0, stream>>>(scores, headmax, partial);
            final_reduce_kernel<<<HEADS, 256, 0, stream>>>(partial, headsum, RED_BLOCKS, 0);

            zero_kernel<<<eltB, 256, 0, stream>>>(attnout, eltN);
            attn_scatter_kernel<<<edgeB16, 256, 0, stream>>>(esrc, edst, scores, headmax,
                                                             headsum, v, attnout);
            gemm64_wmma_kernel<<<gemmB, 128, 0, stream>>>(attnout, pWo, bo, tmp, N_NODES);
            // t = (attn @ Wo.T + bo) @ Wl + bl
            gemm64_wmma_kernel<<<gemmB, 128, 0, stream>>>(tmp, pWlayer, bl,
                                                          branch ? t2 : t1, N_NODES);
        }
        combine_norm_kernel<<<combB, 256, 0, stream>>>(t1, t2, prev, all_emb,
                                                       EMBED * (layer + 1));
    }

    const int gatherB = (3 * BATCH * 64 + 255) / 256;
    gather_out_kernel<<<gatherB, 256, 0, stream>>>(all_emb, users, pos, neg, out);
}
